// MultiHeadLatentAttention_20796231647714
// MI455X (gfx1250) — compile-verified, matrix operand order checked by reading, not run on hardware
//
#include <hip/hip_runtime.h>

// ---------------- problem constants (match reference) ----------------
static constexpr int B_   = 2;
static constexpr int T_   = 2048;
static constexpr int D_   = 2048;
static constexpr int H_   = 16;
static constexpr int HD_  = 128;
static constexpr int LAT_ = 512;
static constexpr int RD_  = 64;
static constexpr int CD_  = HD_ - RD_;   // 64
static constexpr int M_   = B_ * T_;     // 4096
static constexpr int RDP_ = 128;         // k_rope GEMM N padded to 128

typedef __attribute__((ext_vector_type(16))) __bf16 v16bf;
typedef __attribute__((ext_vector_type(8)))  float  v8f;

union Frag16 { v16bf v; unsigned int u[8]; };

__device__ __forceinline__ unsigned short f2bf(float f) {
  unsigned int u = __builtin_bit_cast(unsigned int, f);
  u += 0x7fffu + ((u >> 16) & 1u);           // round-to-nearest-even
  return (unsigned short)(u >> 16);
}

// uint index (bf16 pair) j of a 16-bit A/B WMMA fragment, per CDNA5 layout:
// lanes 0-15 hold K {0..7,16..23}, lanes 16-31 hold K {8..15,24..31}
__device__ __forceinline__ unsigned uidx_of(int j, int hf) {
  return (hf ? 4u : 0u) + (unsigned)(j < 4 ? j : j + 4);
}

// ---------------- f32 -> bf16 conversion ----------------
__global__ void cvt_bf16_kernel(const float* __restrict__ in,
                                unsigned short* __restrict__ out, int n) {
  int i = blockIdx.x * 256 + threadIdx.x;
  if (i < n) out[i] = f2bf(in[i]);
}

// zero-padded row conversion (for Wkr: 64 real rows -> 128 rows)
__global__ void pad_rows_bf16_kernel(const float* __restrict__ src,
                                     unsigned short* __restrict__ dst,
                                     int srcRows, int dstRows, int cols) {
  int idx = blockIdx.x * 256 + threadIdx.x;
  if (idx >= dstRows * cols) return;
  int r = idx / cols;
  dst[idx] = (r < srcRows) ? f2bf(src[idx]) : (unsigned short)0;
}

__global__ void pad_bias_kernel(const float* __restrict__ src,
                                float* __restrict__ dst, int nsrc, int ndst) {
  int i = blockIdx.x * 256 + threadIdx.x;
  if (i < ndst) dst[i] = (i < nsrc) ? src[i] : 0.0f;
}

// ---------------- generic bf16 GEMM: C = A(MxK) * W(NxK)^T + bias ----------------
// block = 256 threads (8 waves). Block tile 128(M) x 128(N), BK = 64.
// Wave tile 64x32 = 4x2 WMMA tiles -> 16 v_wmma per K-slab, 1.5 ds_load_b128/wmma.
__global__ void __launch_bounds__(256)
gemm_bf16_kernel(const unsigned short* __restrict__ A,
                 const unsigned short* __restrict__ W,
                 const float* __restrict__ bias,
                 float* __restrict__ C, int M, int N, int K) {
  __shared__ __align__(16) unsigned int As[128 * 32];   // [row][kpair 0..31] 16 KB
  __shared__ __align__(16) unsigned int Bs[128 * 32];   // [col][kpair 0..31] 16 KB

  const int tid  = threadIdx.x;
  const int wv   = tid >> 5;          // 0..7
  const int lane = tid & 31;
  const int hf   = (lane >> 4) & 1;
  const int l16  = lane & 15;
  const int wRow = (wv & 1) * 64;     // 64-row band in tile
  const int wCol = (wv >> 1) * 32;    // 32-col band in tile
  const int m0   = blockIdx.y * 128;
  const int n0   = blockIdx.x * 128;

  const uint4* Ag4 = (const uint4*)A; // 8 bf16 per uint4
  const uint4* Wg4 = (const uint4*)W;
  const int Ku4 = K >> 3;             // uint4 per row

  v8f acc[4][2] = {};

  for (int k0 = 0; k0 < K; k0 += 64) {
    __syncthreads();
    // stage A tile: 128 rows x 8 uint4 = 1024 uint4 (4 per thread)
#pragma unroll
    for (int i = 0; i < 4; ++i) {
      int e = tid + i * 256;
      int r = e >> 3, c4 = e & 7;
      uint4 g = Ag4[(size_t)(m0 + r) * Ku4 + (k0 >> 3) + c4];
      *(uint4*)&As[e * 4] = g;
    }
    // stage W tile: 128 rows x 8 uint4 = 1024 uint4 (4 per thread)
#pragma unroll
    for (int i = 0; i < 4; ++i) {
      int e = tid + i * 256;
      int r = e >> 3, c4 = e & 7;
      uint4 g = Wg4[(size_t)(n0 + r) * Ku4 + (k0 >> 3) + c4];
      *(uint4*)&Bs[e * 4] = g;
    }
    if (k0 + 64 < K) {  // prefetch next K-slab (global_prefetch_b8)
      __builtin_prefetch((const void*)(Ag4 + (size_t)(m0 + (tid >> 1)) * Ku4 + ((k0 + 64) >> 3)), 0, 1);
      __builtin_prefetch((const void*)(Wg4 + (size_t)(n0 + (tid >> 1)) * Ku4 + ((k0 + 64) >> 3)), 0, 1);
    }
    __syncthreads();

#pragma unroll
    for (int ks = 0; ks < 2; ++ks) {   // two 32-deep WMMA K-steps per slab
      Frag16 a[4], b[2];
#pragma unroll
      for (int i = 0; i < 4; ++i)
#pragma unroll
        for (int j = 0; j < 8; ++j)
          a[i].u[j] = As[(wRow + i * 16 + l16) * 32 + ks * 16 + uidx_of(j, hf)];
#pragma unroll
      for (int i = 0; i < 2; ++i)
#pragma unroll
        for (int j = 0; j < 8; ++j)
          b[i].u[j] = Bs[(wCol + i * 16 + l16) * 32 + ks * 16 + uidx_of(j, hf)];
#pragma unroll
      for (int i = 0; i < 4; ++i)
#pragma unroll
        for (int j = 0; j < 2; ++j)
          acc[i][j] = __builtin_amdgcn_wmma_f32_16x16x32_bf16(false, a[i].v, false, b[j].v,
                                                              (short)0, acc[i][j], false, false);
    }
  }

  // D layout: VGPR r -> M = r + 8*half, N = lane%16
#pragma unroll
  for (int i = 0; i < 4; ++i) {
    const int row = m0 + wRow + i * 16 + hf * 8;
#pragma unroll
    for (int j = 0; j < 2; ++j) {
      const int col = n0 + wCol + j * 16 + l16;
      const float bv = bias[col];
#pragma unroll
      for (int r = 0; r < 8; ++r)
        C[(size_t)(row + r) * N + col] = acc[i][j][r] + bv;
    }
  }
}

// ---------------- RoPE helper (rotate-half, THETA=10000, RD=64) ----------------
__device__ __forceinline__ float rope_val(float x, float xpair, int rd, int t) {
  int i = rd & 31;
  float inv = __powf(10000.0f, -((float)(2 * i)) / 64.0f);
  float ang = (float)t * inv;
  float c = __cosf(ang), s = __sinf(ang);
  return (rd < 32) ? (x * c - xpair * s) : (x * c + xpair * s);
}

// ---------------- q pack: (B*T, H*HD) f32 -> (B,H,T,HD) bf16 with RoPE tail ----
__global__ void pack_q_kernel(const float* __restrict__ qf,
                              unsigned short* __restrict__ qb) {
  int idx = blockIdx.x * 256 + threadIdx.x;
  if (idx >= B_ * H_ * T_ * HD_) return;
  int d = idx % HD_;
  int t = (idx / HD_) % T_;
  int h = (idx / (HD_ * T_)) % H_;
  int b = idx / (HD_ * T_ * H_);
  const float* row = qf + (size_t)(b * T_ + t) * (H_ * HD_) + h * HD_;
  float v = row[d];
  if (d >= CD_) {
    int rd = d - CD_;
    float pair = row[(rd < 32) ? d + 32 : d - 32];
    v = rope_val(v, pair, rd, t);
  }
  qb[idx] = f2bf(v);
}

// ---------------- kv split: (B*T, 2*LAT) f32 -> two (B*T, LAT) bf16 -------------
__global__ void split_kv_kernel(const float* __restrict__ kv,
                                unsigned short* __restrict__ ck,
                                unsigned short* __restrict__ cv) {
  int idx = blockIdx.x * 256 + threadIdx.x;
  if (idx >= M_ * LAT_) return;
  int r = idx / LAT_, c = idx % LAT_;
  const float* row = kv + (size_t)r * (2 * LAT_);
  ck[idx] = f2bf(row[c]);
  cv[idx] = f2bf(row[LAT_ + c]);
}

// ---- k pack: content (B*T,H*CD) + rope (B*T,RDP stride, 64 used) -> (B,H,T,HD) -
__global__ void pack_k_kernel(const float* __restrict__ kc,
                              const float* __restrict__ kr,
                              unsigned short* __restrict__ kb) {
  int idx = blockIdx.x * 256 + threadIdx.x;
  if (idx >= B_ * H_ * T_ * HD_) return;
  int d = idx % HD_;
  int t = (idx / HD_) % T_;
  int h = (idx / (HD_ * T_)) % H_;
  int b = idx / (HD_ * T_ * H_);
  float v;
  if (d < CD_) {
    v = kc[(size_t)(b * T_ + t) * (H_ * CD_) + h * CD_ + d];
  } else {
    int rd = d - CD_;
    const float* row = kr + (size_t)(b * T_ + t) * RDP_;   // stride-128 rows, first 64 valid
    v = rope_val(row[rd], row[(rd < 32) ? rd + 32 : rd - 32], rd, t);
  }
  kb[idx] = f2bf(v);
}

// ---------------- v pack: (B*T, H*HD) f32 -> (B,H,T,HD) bf16 --------------------
__global__ void pack_v_kernel(const float* __restrict__ vf,
                              unsigned short* __restrict__ vb) {
  int idx = blockIdx.x * 256 + threadIdx.x;
  if (idx >= B_ * H_ * T_ * HD_) return;
  int d = idx % HD_;
  int t = (idx / HD_) % T_;
  int h = (idx / (HD_ * T_)) % H_;
  int b = idx / (HD_ * T_ * H_);
  vb[idx] = f2bf(vf[(size_t)(b * T_ + t) * (H_ * HD_) + h * HD_ + d]);
}

// ---------------- causal flash attention (bf16 WMMA, f32 accumulators) ----------
// grid = (T/64, H, B), block = 128 (4 waves, 16 q-rows each)
__global__ void __launch_bounds__(128)
mla_attention_kernel(const unsigned short* __restrict__ Qb,
                     const unsigned short* __restrict__ Kb,
                     const unsigned short* __restrict__ Vb,
                     unsigned short* __restrict__ Ob) {
  __shared__ __align__(16) unsigned int KsU[32 * 64];      // K tile [key][HD/2]     8 KB
  __shared__ __align__(16) unsigned int VsU[128 * 16];     // V transposed [d][key/2] 8 KB
  __shared__ __align__(16) unsigned int PsU[4 * 16 * 16];  // per-wave P tiles        4 KB
  unsigned short* Vs = (unsigned short*)VsU;
  unsigned short* Ps = (unsigned short*)PsU;

  const int tid  = threadIdx.x;
  const int wv   = tid >> 5;
  const int lane = tid & 31;
  const int hf   = (lane >> 4) & 1;
  const int l16  = lane & 15;
  const int qBase = blockIdx.x * 64;
  const int h = blockIdx.y, b = blockIdx.z;
  const size_t headOff = ((size_t)(b * H_ + h)) * T_ * HD_;
  const float scale = 0.08838834764831845f;  // 1/sqrt(128)

  // Q fragments for this wave's 16 rows (16x128 -> 4 x 16x32 A-fragments)
  Frag16 qf[4];
  {
    const unsigned int* Qg = (const unsigned int*)(Qb + headOff);
    int qrow = qBase + wv * 16 + l16;
#pragma unroll
    for (int kk = 0; kk < 4; ++kk)
#pragma unroll
      for (int j = 0; j < 8; ++j)
        qf[kk].u[j] = Qg[(size_t)qrow * 64 + kk * 16 + uidx_of(j, hf)];
  }

  v8f acc[8] = {};
  float mstat[8], lstat[8];
#pragma unroll
  for (int r = 0; r < 8; ++r) { mstat[r] = -1e30f; lstat[r] = 0.0f; }

  const int nK = qBase + 64;                 // causal bound for this block
  for (int kS = 0; kS < nK; kS += 32) {
    __syncthreads();
    // stage K (row-major) and V (transposed to [d][key]) tiles, B128 loads
    const uint4* Kg4 = (const uint4*)(Kb + headOff + (size_t)kS * HD_);
    const uint4* Vg4 = (const uint4*)(Vb + headOff + (size_t)kS * HD_);
#pragma unroll
    for (int i = 0; i < 4; ++i) {
      int e = tid + i * 128;                 // uint4 index, 512 total
      uint4 kq = Kg4[e];
      *(uint4*)&KsU[e * 4] = kq;
      uint4 vq = Vg4[e];
      int key = e >> 4, du4 = e & 15;
      unsigned gs[4] = {vq.x, vq.y, vq.z, vq.w};
#pragma unroll
      for (int t = 0; t < 4; ++t) {
        int d0 = du4 * 8 + t * 2;
        Vs[d0 * 32 + key]       = (unsigned short)(gs[t] & 0xffffu);
        Vs[(d0 + 1) * 32 + key] = (unsigned short)(gs[t] >> 16);
      }
    }
    __syncthreads();

    // S = Q * K^T for two 16-key column tiles
    v8f s0 = {}, s1 = {};
#pragma unroll
    for (int kk = 0; kk < 4; ++kk) {
      Frag16 b0, b1;
#pragma unroll
      for (int j = 0; j < 8; ++j) {
        unsigned ui = kk * 16 + uidx_of(j, hf);
        b0.u[j] = KsU[l16 * 64 + ui];
        b1.u[j] = KsU[(16 + l16) * 64 + ui];
      }
      s0 = __builtin_amdgcn_wmma_f32_16x16x32_bf16(false, qf[kk].v, false, b0.v,
                                                   (short)0, s0, false, false);
      s1 = __builtin_amdgcn_wmma_f32_16x16x32_bf16(false, qf[kk].v, false, b1.v,
                                                   (short)0, s1, false, false);
    }

    // online softmax over the 32 keys; a row lives across 16 lanes of one half
    const int qi0 = qBase + wv * 16 + hf * 8;
#pragma unroll
    for (int r = 0; r < 8; ++r) {
      int qrow = qi0 + r;
      float v0 = (kS + l16      <= qrow) ? s0[r] * scale : -1e30f;
      float v1 = (kS + 16 + l16 <= qrow) ? s1[r] * scale : -1e30f;
      float mx = fmaxf(v0, v1);
      for (int off = 8; off > 0; off >>= 1)
        mx = fmaxf(mx, __shfl_xor(mx, off, 16));
      float newm = fmaxf(mstat[r], mx);
      float p0 = __expf(v0 - newm);
      float p1 = __expf(v1 - newm);
      float rs = p0 + p1;
      for (int off = 8; off > 0; off >>= 1)
        rs += __shfl_xor(rs, off, 16);
      float resc = __expf(mstat[r] - newm);
#pragma unroll
      for (int sl = 0; sl < 8; ++sl) acc[sl][r] *= resc;
      lstat[r] = lstat[r] * resc + rs;
      mstat[r] = newm;
      int prow = r + hf * 8;                 // stage P row-major for A-fragment reload
      Ps[(wv * 16 + prow) * 32 + l16]      = f2bf(p0);
      Ps[(wv * 16 + prow) * 32 + 16 + l16] = f2bf(p1);
    }
    __syncthreads();

    // O += P(16x32) * V(32x128): 8 wmma over HD slices
    Frag16 pa;
#pragma unroll
    for (int j = 0; j < 8; ++j)
      pa.u[j] = PsU[(wv * 16 + l16) * 16 + uidx_of(j, hf)];
#pragma unroll
    for (int sl = 0; sl < 8; ++sl) {
      Frag16 vb;
#pragma unroll
      for (int j = 0; j < 8; ++j)
        vb.u[j] = VsU[(sl * 16 + l16) * 16 + uidx_of(j, hf)];
      acc[sl] = __builtin_amdgcn_wmma_f32_16x16x32_bf16(false, pa.v, false, vb.v,
                                                        (short)0, acc[sl], false, false);
    }
  }

  // normalize + store as (B*T, H*HD) bf16 rows, ready for the Wo GEMM
  const size_t obase = (size_t)(b * T_) * (H_ * HD_);
#pragma unroll
  for (int sl = 0; sl < 8; ++sl)
#pragma unroll
    for (int r = 0; r < 8; ++r) {
      int qrow = qBase + wv * 16 + hf * 8 + r;
      float o = acc[sl][r] / lstat[r];
      Ob[obase + (size_t)qrow * (H_ * HD_) + h * HD_ + sl * 16 + l16] = f2bf(o);
    }
}

// ---------------- host-side orchestration ----------------
extern "C" void kernel_launch(void* const* d_in, const int* in_sizes, int n_in,
                              void* d_out, int out_size, void* d_ws, size_t ws_size,
                              hipStream_t stream) {
  (void)in_sizes; (void)n_in; (void)out_size; (void)ws_size;
  const float* x   = (const float*)d_in[0];
  const float* Wq  = (const float*)d_in[1];
  const float* bq  = (const float*)d_in[2];
  const float* Wkv = (const float*)d_in[3];
  const float* bkv = (const float*)d_in[4];
  const float* Wkr = (const float*)d_in[5];
  const float* bkr = (const float*)d_in[6];
  const float* Wku = (const float*)d_in[7];
  const float* bku = (const float*)d_in[8];
  const float* Wvu = (const float*)d_in[9];
  const float* bvu = (const float*)d_in[10];
  const float* Wo  = (const float*)d_in[11];
  const float* bo  = (const float*)d_in[12];
  float* out = (float*)d_out;

  char* ws = (char*)d_ws;
  size_t off = 0;
  auto alloc = [&](size_t bytes) -> void* {
    void* p = ws + off;
    off += (bytes + 255) & ~(size_t)255;
    return p;
  };

  unsigned short* x_bf    = (unsigned short*)alloc((size_t)M_ * D_ * 2);
  unsigned short* Wq_bf   = (unsigned short*)alloc((size_t)(H_*HD_) * D_ * 2);
  unsigned short* Wkv_bf  = (unsigned short*)alloc((size_t)(2*LAT_) * D_ * 2);
  unsigned short* Wkr_bf  = (unsigned short*)alloc((size_t)RDP_ * D_ * 2);   // zero-padded to 128 rows
  unsigned short* Wku_bf  = (unsigned short*)alloc((size_t)(H_*CD_) * LAT_ * 2);
  unsigned short* Wvu_bf  = (unsigned short*)alloc((size_t)(H_*HD_) * LAT_ * 2);
  unsigned short* Wo_bf   = (unsigned short*)alloc((size_t)D_ * D_ * 2);
  float* bkr_pad = (float*)alloc((size_t)RDP_ * 4);
  float* q_f32   = (float*)alloc((size_t)M_ * (H_*HD_) * 4);
  float* kv_f32  = (float*)alloc((size_t)M_ * (2*LAT_) * 4);
  float* kr_f32  = (float*)alloc((size_t)M_ * RDP_ * 4);                     // stride-128 rows
  float* kc_f32  = (float*)alloc((size_t)M_ * (H_*CD_) * 4);
  float* v_f32   = (float*)alloc((size_t)M_ * (H_*HD_) * 4);
  unsigned short* q_bf  = (unsigned short*)alloc((size_t)M_ * (H_*HD_) * 2);
  unsigned short* k_bf  = (unsigned short*)alloc((size_t)M_ * (H_*HD_) * 2);
  unsigned short* v_bf  = (unsigned short*)alloc((size_t)M_ * (H_*HD_) * 2);
  unsigned short* ck_bf = (unsigned short*)alloc((size_t)M_ * LAT_ * 2);
  unsigned short* cv_bf = (unsigned short*)alloc((size_t)M_ * LAT_ * 2);
  unsigned short* attn_bf = (unsigned short*)alloc((size_t)M_ * (H_*HD_) * 2);

  auto cvt = [&](const float* src, unsigned short* dst, int n) {
    cvt_bf16_kernel<<<dim3((n + 255) / 256), dim3(256), 0, stream>>>(src, dst, n);
  };
  auto gemm = [&](const unsigned short* A, const unsigned short* W,
                  const float* bias, float* C, int M, int N, int K) {
    gemm_bf16_kernel<<<dim3(N / 128, M / 128), dim3(256), 0, stream>>>(A, W, bias, C, M, N, K);
  };

  // bf16 casts (Wkr zero-padded from 64 to 128 output rows)
  cvt(x,   x_bf,   M_ * D_);
  cvt(Wq,  Wq_bf,  H_*HD_ * D_);
  cvt(Wkv, Wkv_bf, 2*LAT_ * D_);
  pad_rows_bf16_kernel<<<dim3((RDP_ * D_ + 255) / 256), dim3(256), 0, stream>>>(
      Wkr, Wkr_bf, RD_, RDP_, D_);
  pad_bias_kernel<<<dim3(1), dim3(256), 0, stream>>>(bkr, bkr_pad, RD_, RDP_);
  cvt(Wku, Wku_bf, H_*CD_ * LAT_);
  cvt(Wvu, Wvu_bf, H_*HD_ * LAT_);
  cvt(Wo,  Wo_bf,  D_ * D_);

  // projections
  gemm(x_bf, Wq_bf,  bq,      q_f32,  M_, H_*HD_, D_);
  gemm(x_bf, Wkv_bf, bkv,     kv_f32, M_, 2*LAT_, D_);
  gemm(x_bf, Wkr_bf, bkr_pad, kr_f32, M_, RDP_,   D_);

  // pack/RoPE q, split latent kv
  const int nq = B_ * H_ * T_ * HD_;
  pack_q_kernel<<<dim3((nq + 255) / 256), dim3(256), 0, stream>>>(q_f32, q_bf);
  const int nkv = M_ * LAT_;
  split_kv_kernel<<<dim3((nkv + 255) / 256), dim3(256), 0, stream>>>(kv_f32, ck_bf, cv_bf);

  // up-projections
  gemm(ck_bf, Wku_bf, bku, kc_f32, M_, H_*CD_, LAT_);
  gemm(cv_bf, Wvu_bf, bvu, v_f32,  M_, H_*HD_, LAT_);

  // pack k (content + RoPE'd shared rope key) and v into (B,H,T,HD) bf16
  pack_k_kernel<<<dim3((nq + 255) / 256), dim3(256), 0, stream>>>(kc_f32, kr_f32, k_bf);
  pack_v_kernel<<<dim3((nq + 255) / 256), dim3(256), 0, stream>>>(v_f32, v_bf);

  // causal flash attention
  mla_attention_kernel<<<dim3(T_ / 64, H_, B_), dim3(128), 0, stream>>>(q_bf, k_bf, v_bf, attn_bf);

  // output projection -> f32 result
  gemm(attn_bf, Wo_bf, bo, out, M_, D_, D_);
}
